// ScaledDotAttention_36550171689565
// MI455X (gfx1250) — matrix operand order repaired and checked by reading, not compiled, not run on hardware
//
#include <hip/hip_runtime.h>

typedef __bf16 bf16_t;
typedef __attribute__((ext_vector_type(16))) __bf16 v16bf;
typedef __attribute__((ext_vector_type(8)))  float  v8f;
typedef __attribute__((ext_vector_type(4)))  unsigned int u32x4;
typedef __attribute__((ext_vector_type(8)))  int i32x8;
typedef __attribute__((ext_vector_type(4)))  int i32x4;

#define HID   1024
#define BATCH 16
#define SEQ   4096
#define KQ    64
#define LDT   40      // padded LDS row stride in bf16 elements (32 data + 8 pad)

template<bool C, typename T, typename F> struct cond_t { typedef T type; };
template<typename T, typename F> struct cond_t<false, T, F> { typedef F type; };

#if defined(__gfx1250__) && __has_builtin(__builtin_amdgcn_tensor_load_to_lds)
#define HAVE_TDM 1
#else
#define HAVE_TDM 0
#endif

// ---------------------------------------------------------------------------
// TDM: DMA a (rows x 32) bf16 tile (row stride = stride_elems) from global
// into LDS at byte offset lds_off, padding each 64B row by 16B so rows land
// at the 40-half stride the WMMA fragment gathers expect. D# per ISA ch.8.
// Must be called under a wave-uniform branch (one wave per workgroup).
// ---------------------------------------------------------------------------
__device__ __forceinline__ void tdm_load_tile_bf16(unsigned int lds_off,
                                                   const bf16_t* gptr,
                                                   unsigned int rows,
                                                   unsigned int stride_elems)
{
#if HAVE_TDM
    unsigned long long ga = (unsigned long long)(uintptr_t)gptr;
    u32x4 g0;
    g0[0] = 1u;                                           // count=1 valid descriptor
    g0[1] = lds_off;                                      // lds_addr (bytes)
    g0[2] = (unsigned int)(ga & 0xffffffffull);           // global_addr[31:0]
    g0[3] = (unsigned int)((ga >> 32) & 0x1ffffffull)     // global_addr[56:32]
          | (2u << 30);                                   // type = 2 ("image")

    i32x8 g1;
    // data_size=2B (code 1), pad_enable=1, pad_interval=3 (16 DWORDs = 64B),
    // pad_amount=3 (4 DWORDs = 16B) -> LDS row stride 40 halfs
    g1[0] = (1 << 16) | (1 << 20) | (3 << 22) | (3 << 25);
    const unsigned int d0 = 32u, d1 = rows;
    g1[1] = (int)((d0 & 0xffffu) << 16);                  // tensor_dim0[15:0]
    g1[2] = (int)(((d0 >> 16) & 0xffffu) | ((d1 & 0xffffu) << 16)); // dim0 hi | dim1 lo
    g1[3] = (int)(((d1 >> 16) & 0xffffu) | (32u << 16));  // dim1 hi | tile_dim0=32
    g1[4] = (int)(rows & 0xffffu);                        // tile_dim1=rows, tile_dim2=0
    g1[5] = (int)stride_elems;                            // tensor_dim0_stride[31:0]
    g1[6] = 0;                                            // stride hi | dim1_stride lo
    g1[7] = 0;

    i32x4 z4 = {0, 0, 0, 0};
#if __clang_major__ >= 23
    i32x8 z8 = {0, 0, 0, 0, 0, 0, 0, 0};
    __builtin_amdgcn_tensor_load_to_lds(g0, g1, z4, z4, z8, 0);
#else
    __builtin_amdgcn_tensor_load_to_lds(g0, g1, z4, z4, 0);
#endif
#else
    (void)lds_off; (void)gptr; (void)rows; (void)stride_elems;
#endif
}

// --------------- WMMA fragment helpers (wave32 layouts, ISA 7.12.2) --------
// A 16x32 bf16: lanes 0-15: M=lane, K={0..7,16..23}; lanes 16-31: K={8..15,24..31}
__device__ __forceinline__ v16bf frag_a(const bf16_t* tile, int wave, int lane)
{
    v16bf af;
    const int m    = wave * 16 + (lane & 15);
    const int half = lane >> 4;
#pragma unroll
    for (int e = 0; e < 16; ++e) {
        const int k = (e >> 3) * 16 + half * 8 + (e & 7);
        af[e] = tile[m * LDT + k];
    }
    return af;
}
// B 32x16 bf16 from tile stored [n][k]: lanes 0-15 K={0..7,16..23}, lanes 16-31 K={8..15,24..31}
__device__ __forceinline__ v16bf frag_b(const bf16_t* tile, int nt, int lane)
{
    v16bf bfb;
    const int n    = nt * 16 + (lane & 15);
    const int half = lane >> 4;
#pragma unroll
    for (int e = 0; e < 16; ++e)
        bfb[e] = tile[n * LDT + (e >> 3) * 16 + half * 8 + (e & 7)];
    return bfb;
}

// ---------------------------------------------------------------------------
// GEMM: C[M,N] = scale * (A[M,K] @ B[N,K]^T) + bias[N]   (fp32 in, bf16 out)
// Block = 128 threads (4 waves), tile 64x64, K-step 32. Used for Q/K/V proj.
// ---------------------------------------------------------------------------
template<bool HAS_BIAS>
__global__ void __launch_bounds__(128)
gemm_abt_wmma(const float* __restrict__ A, const float* __restrict__ B,
              const float* __restrict__ bias, bf16_t* __restrict__ C,
              int K, int lda, int ldb, int ldc, float scale)
{
    __shared__ bf16_t Asm[64 * LDT];
    __shared__ bf16_t Bsm[64 * LDT];

    const int tid  = threadIdx.x;
    const int lane = tid & 31;
    const int wave = tid >> 5;
    const long m0 = (long)blockIdx.x * 64;
    const long n0 = (long)blockIdx.y * 64;

    v8f zero = {0.f,0.f,0.f,0.f,0.f,0.f,0.f,0.f};
    v8f acc[4] = {zero, zero, zero, zero};

    const int lr = tid >> 1;            // 0..63
    const int lc = (tid & 1) * 16;      // 0 or 16

    for (int k0 = 0; k0 < K; k0 += 32) {
        const float* pa = A + (m0 + lr) * (long)lda + k0 + lc;
        const float* pb = B + (n0 + lr) * (long)ldb + k0 + lc;
        __builtin_prefetch(pa + 32, 0, 0);   // -> global_prefetch_b8
        __builtin_prefetch(pb + 32, 0, 0);
#pragma unroll
        for (int c = 0; c < 16; ++c) Asm[lr * LDT + lc + c] = (bf16_t)pa[c];
#pragma unroll
        for (int c = 0; c < 16; ++c) Bsm[lr * LDT + lc + c] = (bf16_t)pb[c];
        __syncthreads();

        const v16bf af = frag_a(Asm, wave, lane);
#pragma unroll
        for (int nt = 0; nt < 4; ++nt)
            acc[nt] = __builtin_amdgcn_wmma_f32_16x16x32_bf16(
                false, af, false, frag_b(Bsm, nt, lane), (short)0, acc[nt], false, false);
        __syncthreads();
    }

    const int nlane = lane & 15;
    const int moff  = (lane >> 4) * 8;
#pragma unroll
    for (int nt = 0; nt < 4; ++nt) {
        const long n = n0 + nt * 16 + nlane;
        const float bv = HAS_BIAS ? bias[n] : 0.f;
#pragma unroll
        for (int j = 0; j < 8; ++j) {
            const long m = m0 + wave * 16 + moff + j;
            C[m * (long)ldc + n] = (bf16_t)(acc[nt][j] * scale + bv);
        }
    }
}

// ---------------------------------------------------------------------------
// scores[b] = scale * k[b](S x H) @ q[b](KQ x H)^T   (bf16 in, f32 out)
// TDM double-buffered tile staging: wave 0 DMAs the next K-tile pair while
// all 4 waves run WMMA on the current pair. grid = (S/64, 1, B).
// ---------------------------------------------------------------------------
__global__ void __launch_bounds__(128)
gemm_scores_tdm(const bf16_t* __restrict__ kb, const bf16_t* __restrict__ qb,
                float* __restrict__ out, float scale)
{
    __shared__ bf16_t Asm[2][64 * LDT];
    __shared__ bf16_t Bsm[2][64 * LDT];

    const int tid  = threadIdx.x;
    const int lane = tid & 31;
    const int wave = tid >> 5;
    const int b    = blockIdx.z;
    const long m0  = (long)blockIdx.x * 64;

    const bf16_t* A = kb + (long)b * SEQ * HID + m0 * HID;  // 64 rows of k
    const bf16_t* B = qb + (long)b * KQ * HID;              // all 64 rows of q

    v8f zero = {0.f,0.f,0.f,0.f,0.f,0.f,0.f,0.f};
    v8f acc[4] = {zero, zero, zero, zero};

#if HAVE_TDM
    if (wave == 0) {
        tdm_load_tile_bf16((unsigned int)(uintptr_t)&Asm[0][0], A, 64, HID);
        tdm_load_tile_bf16((unsigned int)(uintptr_t)&Bsm[0][0], B, 64, HID);
    }
#endif
    const int NT = HID / 32;
    for (int t = 0; t < NT; ++t) {
        const int cur = t & 1;
#if HAVE_TDM
        if (wave == 0) {
            if (t + 1 < NT) {   // prefetch next tile pair into the other buffer
                tdm_load_tile_bf16((unsigned int)(uintptr_t)&Asm[cur ^ 1][0],
                                   A + (t + 1) * 32, 64, HID);
                tdm_load_tile_bf16((unsigned int)(uintptr_t)&Bsm[cur ^ 1][0],
                                   B + (t + 1) * 32, 64, HID);
                __builtin_amdgcn_s_wait_tensorcnt(2);  // current pair retired
            } else {
                __builtin_amdgcn_s_wait_tensorcnt(0);
            }
        }
#else
        {   // fallback: cooperative vector loads
            const int lr = tid >> 1, lc = (tid & 1) * 16, k0 = t * 32;
            const bf16_t* pa = A + (long)lr * HID + k0 + lc;
            const bf16_t* pb = B + (long)lr * HID + k0 + lc;
#pragma unroll
            for (int c = 0; c < 16; ++c) Asm[cur][lr * LDT + lc + c] = pa[c];
#pragma unroll
            for (int c = 0; c < 16; ++c) Bsm[cur][lr * LDT + lc + c] = pb[c];
        }
#endif
        __syncthreads();
        const v16bf af = frag_a(Asm[cur], wave, lane);
#pragma unroll
        for (int nt = 0; nt < 4; ++nt)
            acc[nt] = __builtin_amdgcn_wmma_f32_16x16x32_bf16(
                false, af, false, frag_b(Bsm[cur], nt, lane), (short)0, acc[nt], false, false);
        __syncthreads();
    }

    float* Cb = out + (long)b * SEQ * KQ;
    const int nlane = lane & 15;
    const int moff  = (lane >> 4) * 8;
#pragma unroll
    for (int nt = 0; nt < 4; ++nt) {
        const long n = nt * 16 + nlane;
#pragma unroll
        for (int j = 0; j < 8; ++j) {
            const long m = m0 + wave * 16 + moff + j;
            Cb[m * KQ + n] = acc[nt][j] * scale;
        }
    }
}

// ---------------------------------------------------------------------------
// context[b] = attnT[b](KQ x S) @ v[b](S x H)    (bf16 in, f32 out)
// B (v) tile is stored TRANSPOSED into LDS ([n][k], stride 40) so that every
// fragment gather is K-contiguous -> ds_load_b128 instead of u16 chains.
// grid = (1, H/64, B).
// ---------------------------------------------------------------------------
__global__ void __launch_bounds__(128)
gemm_ab_wmma(const bf16_t* __restrict__ Aall, const bf16_t* __restrict__ Ball,
             float* __restrict__ Call, int K, int lda, int ldb, int ldc,
             long sA, long sB, long sC)
{
    __shared__ bf16_t Asm[64 * LDT];
    __shared__ bf16_t Bsm[64 * LDT];   // transposed: [n][k]

    const bf16_t* A = Aall + (long)blockIdx.z * sA;
    const bf16_t* B = Ball + (long)blockIdx.z * sB;

    const int tid  = threadIdx.x;
    const int lane = tid & 31;
    const int wave = tid >> 5;
    const long m0 = (long)blockIdx.x * 64;
    const long n0 = (long)blockIdx.y * 64;

    v8f zero = {0.f,0.f,0.f,0.f,0.f,0.f,0.f,0.f};
    v8f acc[4] = {zero, zero, zero, zero};

    const int ar = tid >> 1;            // A tile: 64 rows x 32 k
    const int ac = (tid & 1) * 16;
    const int br = tid >> 2;            // B tile: 32 k-rows x 64 n (global)
    const int bc = (tid & 3) * 16;

    for (int k0 = 0; k0 < K; k0 += 32) {
        const bf16_t* pa = A + (m0 + ar) * (long)lda + k0 + ac;
        const bf16_t* pb = B + (long)(k0 + br) * (long)ldb + n0 + bc;
        __builtin_prefetch(pa + 32, 0, 0);
        __builtin_prefetch(pb + 32 * ldb, 0, 0);
#pragma unroll
        for (int c = 0; c < 16; ++c) Asm[ar * LDT + ac + c] = pa[c];
        // coalesced global read, transposed scatter into LDS
#pragma unroll
        for (int c = 0; c < 16; ++c) Bsm[(bc + c) * LDT + br] = pb[c];
        __syncthreads();

        const v16bf af = frag_a(Asm, wave, lane);
#pragma unroll
        for (int nt = 0; nt < 4; ++nt)
            acc[nt] = __builtin_amdgcn_wmma_f32_16x16x32_bf16(
                false, af, false, frag_b(Bsm, nt, lane), (short)0, acc[nt], false, false);
        __syncthreads();
    }

    const int nlane = lane & 15;
    const int moff  = (lane >> 4) * 8;
#pragma unroll
    for (int nt = 0; nt < 4; ++nt) {
        const long n = n0 + nt * 16 + nlane;
#pragma unroll
        for (int j = 0; j < 8; ++j) {
            const long m = m0 + wave * 16 + moff + j;
            Call[(long)blockIdx.z * sC + m * (long)ldc + n] = acc[nt][j];
        }
    }
}

// ---------------------------------------------------------------------------
// Softmax over S (axis=1 of (B,S,Kq)) pass 1: per-(b,qi) column max and sum.
// ---------------------------------------------------------------------------
__global__ void __launch_bounds__(256)
softmax_reduce(const float* __restrict__ scores,
               float* __restrict__ cmax, float* __restrict__ csum)
{
    __shared__ float red[256];
    const int col = blockIdx.x;             // b*KQ + qi
    const int b  = col >> 6;
    const int qi = col & 63;
    const float* p = scores + (long)b * SEQ * KQ + qi;

    float vals[16];
    float mx = -3.402823e38f;
#pragma unroll
    for (int i = 0; i < 16; ++i) {
        vals[i] = p[(long)(threadIdx.x + i * 256) * KQ];
        mx = fmaxf(mx, vals[i]);
    }
    red[threadIdx.x] = mx;
    __syncthreads();
    for (int s = 128; s > 0; s >>= 1) {
        if (threadIdx.x < s) red[threadIdx.x] = fmaxf(red[threadIdx.x], red[threadIdx.x + s]);
        __syncthreads();
    }
    mx = red[0];
    __syncthreads();

    float sum = 0.f;
#pragma unroll
    for (int i = 0; i < 16; ++i) sum += __expf(vals[i] - mx);
    red[threadIdx.x] = sum;
    __syncthreads();
    for (int s = 128; s > 0; s >>= 1) {
        if (threadIdx.x < s) red[threadIdx.x] += red[threadIdx.x + s];
        __syncthreads();
    }
    if (threadIdx.x == 0) { cmax[col] = mx; csum[col] = red[0]; }
}

// ---------------------------------------------------------------------------
// Softmax pass 2: normalize in place (fp32 -> d_out) and emit attn^T in bf16.
// ---------------------------------------------------------------------------
__global__ void __launch_bounds__(256)
softmax_norm(float* __restrict__ attn, const float* __restrict__ cmax,
             const float* __restrict__ csum, bf16_t* __restrict__ attnT)
{
    const long idx = (long)blockIdx.x * 256 + threadIdx.x;   // over B*SEQ*KQ
    const int qi = (int)(idx & 63);
    const int s  = (int)((idx >> 6) & 4095);
    const int b  = (int)(idx >> 18);
    const int col = b * 64 + qi;
    const float w = __expf(attn[idx] - cmax[col]) / csum[col];
    attn[idx] = w;
    attnT[((long)b * KQ + qi) * SEQ + s] = (bf16_t)w;
}

// ---------------------------------------------------------------------------
extern "C" void kernel_launch(void* const* d_in, const int* in_sizes, int n_in,
                              void* d_out, int out_size, void* d_ws, size_t ws_size,
                              hipStream_t stream)
{
    const float* queries = (const float*)d_in[0];   // (B, KQ, H)
    const float* keys    = (const float*)d_in[1];   // (B, S, H)
    const float* values  = (const float*)d_in[2];   // (B, S, H)
    const float* Wq = (const float*)d_in[3];
    const float* bq = (const float*)d_in[4];
    const float* Wk = (const float*)d_in[5];
    const float* bk = (const float*)d_in[6];
    const float* Wv = (const float*)d_in[7];
    const float* bv = (const float*)d_in[8];

    // workspace layout (bytes)
    char* ws = (char*)d_ws;
    bf16_t* qb    = (bf16_t*)(ws);                   //   2 MB (B*KQ x H)
    bf16_t* kb    = (bf16_t*)(ws + (2l   << 20));    // 128 MB (B*S x H)
    bf16_t* vb    = (bf16_t*)(ws + (130l << 20));    // 128 MB
    bf16_t* attnT = (bf16_t*)(ws + (258l << 20));    //   8 MB (B, KQ, S)
    float*  cmax  = (float*) (ws + (266l << 20));
    float*  csum  = cmax + BATCH * KQ;

    float* ctx_out  = (float*)d_out;                             // (B, KQ, H)
    float* attn_out = ctx_out + (long)BATCH * KQ * HID;          // (B, S, KQ)

    const float scale = 0.03125f;  // 1/sqrt(1024)

    // 1) q = queries @ Wq^T + bq  -> bf16   (M = B*KQ = 1024)
    gemm_abt_wmma<true><<<dim3(16, 16, 1), 128, 0, stream>>>(
        queries, Wq, bq, qb, HID, HID, HID, HID, 1.f);
    // 2) k = keys @ Wk^T + bk    -> bf16   (M = B*S = 65536)
    gemm_abt_wmma<true><<<dim3(1024, 16, 1), 128, 0, stream>>>(
        keys, Wk, bk, kb, HID, HID, HID, HID, 1.f);
    // 3) v = values @ Wv^T + bv  -> bf16
    gemm_abt_wmma<true><<<dim3(1024, 16, 1), 128, 0, stream>>>(
        values, Wv, bv, vb, HID, HID, HID, HID, 1.f);
    // 4) scores[b] = k[b] @ q[b]^T * scale -> fp32 (B,S,KQ) in d_out (TDM path)
    gemm_scores_tdm<<<dim3(SEQ / 64, 1, BATCH), 128, 0, stream>>>(
        kb, qb, attn_out, scale);
    // 5) softmax over S: column max & sum(exp)
    softmax_reduce<<<dim3(BATCH * KQ), 256, 0, stream>>>(attn_out, cmax, csum);
    // 6) normalize -> fp32 weights (d_out) + bf16 attn^T (ws)
    softmax_norm<<<dim3((BATCH * SEQ * KQ) / 256), 256, 0, stream>>>(
        attn_out, cmax, csum, attnT);
    // 7) context[b] = attnT[b] (KQ x S) @ v[b] (S x H) -> fp32 (B, KQ, H)
    gemm_ab_wmma<<<dim3(1, HID / 64, BATCH), 128, 0, stream>>>(
        attnT, vb, ctx_out, SEQ, SEQ, HID, HID,
        (long)KQ * SEQ, (long)SEQ * HID, (long)KQ * HID);
}
